// GqaAndSwa_51634096832856
// MI455X (gfx1250) — matrix operand order repaired and checked by reading, not compiled, not run on hardware
//
#include <hip/hip_runtime.h>
#include <math.h>

// ---------------------------------------------------------------------------
// GQA + sliding-window attention block for MI455X (gfx1250, wave32, WMMA).
// Pipeline: fp32->bf16 cast | 3x WMMA GEMM (QKV proj, async-LDS double
// buffered) | RoPE | flash attention (v_wmma_f32_16x16x32_bf16 for QK^T and
// PV) | WMMA GEMM (Wo proj).
// Reference's _split_kv reshape chain maps head h of K/V to feature columns
// [128h, 128h+128) -- identical to Q -- so attention is per-head 1:1.
// ---------------------------------------------------------------------------

typedef __attribute__((ext_vector_type(16))) __bf16        v16bf;
typedef __attribute__((ext_vector_type(8)))  float         v8f;
typedef __attribute__((ext_vector_type(4)))  unsigned int  u32x4;

union FragU { u32x4 q[2]; v16bf v; };   // 32 bytes = 16 bf16 = one WMMA operand

static constexpr int S   = 2048;   // sequence length
static constexpr int DM  = 4096;   // model dim
static constexpr int H   = 32;     // heads
static constexpr int HD  = 128;    // head dim
static constexpr int WIN = 1024;   // sliding window

__device__ __forceinline__ unsigned short f2bf(float f) {
  unsigned int u = __float_as_uint(f);                 // round-to-nearest-even
  return (unsigned short)((u + 0x7FFFu + ((u >> 16) & 1u)) >> 16);
}
__device__ __forceinline__ float bf2f(unsigned short u) {
  return __uint_as_float(((unsigned int)u) << 16);
}
__device__ __forceinline__ v8f wmma_bf16(v16bf a, v16bf b, v8f c) {
  // D(16x16,f32) = A(16x32,bf16) x B(32x16,bf16) + C
  return __builtin_amdgcn_wmma_f32_16x16x32_bf16(false, a, false, b,
                                                 (short)0, c, false, false);
}

// Async global->LDS 16B copy (CDNA5 GLOBAL_LOAD_ASYNC_TO_LDS_B128, GV mode).
// lds_off: wave-relative LDS byte address (= low 32 bits of generic pointer,
// per the flat-aperture mapping). Tracked by ASYNCcnt.
__device__ __forceinline__ void async_g2l_b128(unsigned lds_off,
                                               const void* gaddr) {
  asm volatile("global_load_async_to_lds_b128 %0, %1, off"
               :
               : "v"(lds_off), "v"((unsigned long long)gaddr)
               : "memory");
}
__device__ __forceinline__ void wait_asynccnt0() {
  asm volatile("s_wait_asynccnt 0" ::: "memory");
}
__device__ __forceinline__ unsigned lds_off32(const void* p) {
  return (unsigned)(size_t)p;
}

// --------------------------- fp32 -> bf16 cast -----------------------------
__global__ void cast_f32_bf16(const float* __restrict__ src,
                              unsigned short* __restrict__ dst, int n) {
  int i = blockIdx.x * blockDim.x + threadIdx.x;
  if (i < n) dst[i] = f2bf(src[i]);
}

// ---------------- WMMA GEMM: C[M,N] = A[M,K] * B[N,K]^T --------------------
// A,B bf16 row-major. 64x64 tile per 128-thread (4-wave) workgroup. K staged
// through double-buffered LDS in 32-wide steps with GLOBAL_LOAD_ASYNC_TO_LDS
// overlapping the WMMA work on the current tile. mode selects the epilogue:
//   0: bf16 out[h][s][d]   (Q/K head-major)        h=n>>7, d=n&127
//   1: bf16 out[h][d][s]   (V transposed)
//   2: f32  out[m][n]      (final projection to d_out)
__global__ __launch_bounds__(128)
void gemm_bt_wmma(const unsigned short* __restrict__ A,
                  const unsigned short* __restrict__ B,
                  void* __restrict__ Cout, int mode) {
  __shared__ unsigned short As[2][64 * 32];
  __shared__ unsigned short Bs[2][64 * 32];
  const int K = DM;
  const int m0 = blockIdx.x * 64;
  const int n0 = blockIdx.y * 64;
  const int t    = threadIdx.x;
  const int lane = t & 31;
  const int wave = t >> 5;
  const int lh = lane >> 4;       // lane half (C layout: rows M and M+8)
  const int lm = lane & 15;
  const int bA = lh * 16;         // A-frag contiguous K base
  const int kB = lh * 16;         // B-frag contiguous K base

  // This thread's two staging chunks (16B each) of the 64x32 bf16 tiles.
  const int c0 = t, c1 = t + 128;
  const int r0 = c0 >> 2, o0 = (c0 & 3) * 8;
  const int r1 = c1 >> 2, o1 = (c1 & 3) * 8;
  const size_t ga0 = (size_t)(m0 + r0) * K + o0;
  const size_t ga1 = (size_t)(m0 + r1) * K + o1;
  const size_t gb0 = (size_t)(n0 + r0) * K + o0;
  const size_t gb1 = (size_t)(n0 + r1) * K + o1;

  v8f z = {0.f, 0.f, 0.f, 0.f, 0.f, 0.f, 0.f, 0.f};
  v8f acc[4];
  for (int j = 0; j < 4; ++j) acc[j] = z;

  // prologue: stage K-tile 0 into buffer 0
  async_g2l_b128(lds_off32(&As[0][c0 * 8]), &A[ga0]);
  async_g2l_b128(lds_off32(&As[0][c1 * 8]), &A[ga1]);
  async_g2l_b128(lds_off32(&Bs[0][c0 * 8]), &B[gb0]);
  async_g2l_b128(lds_off32(&Bs[0][c1 * 8]), &B[gb1]);
  wait_asynccnt0();
  __syncthreads();

  for (int k0 = 0; k0 < K; k0 += 32) {
    const int buf = (k0 >> 5) & 1;
    if (k0 + 32 < K) {   // async-stage next K tile while computing this one
      const int nb = buf ^ 1, kn = k0 + 32;
      async_g2l_b128(lds_off32(&As[nb][c0 * 8]), &A[ga0 + kn]);
      async_g2l_b128(lds_off32(&As[nb][c1 * 8]), &A[ga1 + kn]);
      async_g2l_b128(lds_off32(&Bs[nb][c0 * 8]), &B[gb0 + kn]);
      async_g2l_b128(lds_off32(&Bs[nb][c1 * 8]), &B[gb1 + kn]);
    }
    FragU fa;
    const int rowA = wave * 16 + lm;
    fa.q[0] = *(const u32x4*)&As[buf][rowA * 32 + bA];
    fa.q[1] = *(const u32x4*)&As[buf][rowA * 32 + bA + 8];
#pragma unroll
    for (int j = 0; j < 4; ++j) {
      FragU fb;
      const int rowB = j * 16 + lm;
      fb.q[0] = *(const u32x4*)&Bs[buf][rowB * 32 + kB];
      fb.q[1] = *(const u32x4*)&Bs[buf][rowB * 32 + kB + 8];
      acc[j] = wmma_bf16(fa.v, fb.v, acc[j]);
    }
    wait_asynccnt0();     // next tile fully landed in LDS
    __syncthreads();      // all waves done reading current tile
  }
#pragma unroll
  for (int j = 0; j < 4; ++j) {
#pragma unroll
    for (int g = 0; g < 8; ++g) {
      const int m = m0 + wave * 16 + g + 8 * lh;  // C layout row
      const int n = n0 + j * 16 + lm;             // C layout col
      const float val = acc[j][g];
      if (mode == 0) {
        ((unsigned short*)Cout)[((size_t)(n >> 7) * S + m) * HD + (n & 127)] =
            f2bf(val);
      } else if (mode == 1) {
        ((unsigned short*)Cout)[((size_t)(n >> 7) * HD + (n & 127)) * S + m] =
            f2bf(val);
      } else {
        ((float*)Cout)[(size_t)m * DM + n] = val;
      }
    }
  }
}

// ------------------------------- RoPE --------------------------------------
// In-place on Q,K [h][s][128]; each thread owns the (d, d+64) pair.
__global__ void rope_kernel(unsigned short* __restrict__ Qh,
                            unsigned short* __restrict__ Kh) {
  int idx = blockIdx.x * blockDim.x + threadIdx.x;   // h*131072 + s*64 + dh
  if (idx >= H * S * 64) return;
  const int dh = idx & 63;
  const int s  = (idx >> 6) & (S - 1);
  const int h  = idx >> 17;
  const float freq = __expf(-(float)dh * (1.0f / 64.0f) * 9.2103403719761836f);
  float sn, cs;
  __sincosf((float)s * freq, &sn, &cs);
  const size_t base = ((size_t)h * S + s) * HD;
  float q1 = bf2f(Qh[base + dh]), q2 = bf2f(Qh[base + dh + 64]);
  Qh[base + dh]      = f2bf(q1 * cs - q2 * sn);
  Qh[base + dh + 64] = f2bf(q2 * cs + q1 * sn);
  float k1 = bf2f(Kh[base + dh]), k2 = bf2f(Kh[base + dh + 64]);
  Kh[base + dh]      = f2bf(k1 * cs - k2 * sn);
  Kh[base + dh + 64] = f2bf(k2 * cs + k1 * sn);
}

// ------------------------ flash attention (1 wave) -------------------------
// One wave per (head, 16-row q block). Online softmax over 32-key blocks
// restricted to the causal sliding window. P converted C-layout -> A-layout
// via a 1KB LDS round trip.
__global__ __launch_bounds__(32)
void attn_kernel(const unsigned short* __restrict__ Qh,
                 const unsigned short* __restrict__ Kh,
                 const unsigned short* __restrict__ Vt,
                 unsigned short* __restrict__ ctx) {
  __shared__ unsigned short Ps[16 * 32];
  const int h  = blockIdx.x >> 7;
  const int q0 = (blockIdx.x & 127) * 16;
  const int lane = threadIdx.x;
  const int lh = lane >> 4, lm = lane & 15;
  const int bA = lh * 16, kB = lh * 16;

  const unsigned short* Qb = Qh + (size_t)h * S * HD;
  const unsigned short* Kb = Kh + (size_t)h * S * HD;
  const unsigned short* Vb = Vt + (size_t)h * HD * S;

  FragU qf[4];                       // Q tile 16x128 as 4 A-fragments
#pragma unroll
  for (int c = 0; c < 4; ++c) {
    const unsigned short* p = Qb + (size_t)(q0 + lm) * HD + c * 32;
    qf[c].q[0] = *(const u32x4*)(p + bA);
    qf[c].q[1] = *(const u32x4*)(p + bA + 8);
  }

  v8f z = {0.f, 0.f, 0.f, 0.f, 0.f, 0.f, 0.f, 0.f};
  v8f o[8];
#pragma unroll
  for (int d = 0; d < 8; ++d) o[d] = z;
  float mrow[8], lrow[8];
#pragma unroll
  for (int g = 0; g < 8; ++g) { mrow[g] = -1e30f; lrow[g] = 0.f; }

  int ks = q0 - (WIN - 1);
  if (ks < 0) ks = 0;
  ks &= ~31;
  const float scale = 0.088388347648318447f;   // 1/sqrt(128)

  for (int kb = ks; kb <= q0 + 15; kb += 32) {
    // prefetch next key block (global_prefetch_b8)
    if (kb + 32 <= q0 + 15) {
      __builtin_prefetch(Kb + (size_t)(kb + 32 + lm) * HD, 0, 1);
    }
    // ---- S = Q K^T for two 16-key sub-tiles (8 WMMAs) ----
    v8f s0 = z, s1 = z;
#pragma unroll
    for (int c = 0; c < 4; ++c) {
      const unsigned short* kp = Kb + (size_t)(kb + lm) * HD + c * 32 + kB;
      FragU f0; f0.q[0] = *(const u32x4*)kp; f0.q[1] = *(const u32x4*)(kp + 8);
      s0 = wmma_bf16(qf[c].v, f0.v, s0);
      const unsigned short* kp1 = kp + 16 * HD;
      FragU f1; f1.q[0] = *(const u32x4*)kp1; f1.q[1] = *(const u32x4*)(kp1 + 8);
      s1 = wmma_bf16(qf[c].v, f1.v, s1);
    }
    // ---- mask + online softmax (branch-free; EXEC stays full) ----
#pragma unroll
    for (int g = 0; g < 8; ++g) {
      const int m  = q0 + g + 8 * lh;
      const int ka = kb + lm, kc = kb + 16 + lm;
      float a = (ka <= m && ka + WIN > m) ? s0[g] * scale : -1e30f;
      float b = (kc <= m && kc + WIN > m) ? s1[g] * scale : -1e30f;
      float rm = fmaxf(a, b);
      rm = fmaxf(rm, __shfl_xor(rm, 1, 16));
      rm = fmaxf(rm, __shfl_xor(rm, 2, 16));
      rm = fmaxf(rm, __shfl_xor(rm, 4, 16));
      rm = fmaxf(rm, __shfl_xor(rm, 8, 16));
      const float mn  = fmaxf(mrow[g], rm);
      const float mns = (mn < -1e29f) ? 0.f : mn;   // fully-masked block guard
      const float corr = __expf(mrow[g] - mns);
      const float pa = __expf(a - mns);
      const float pb = __expf(b - mns);
      float rs = pa + pb;
      rs += __shfl_xor(rs, 1, 16);
      rs += __shfl_xor(rs, 2, 16);
      rs += __shfl_xor(rs, 4, 16);
      rs += __shfl_xor(rs, 8, 16);
      lrow[g] = lrow[g] * corr + rs;
      mrow[g] = mn;
#pragma unroll
      for (int d = 0; d < 8; ++d) o[d][g] = o[d][g] * corr;
      Ps[(g + 8 * lh) * 32 + lm]      = f2bf(pa);   // C layout -> LDS [m][k]
      Ps[(g + 8 * lh) * 32 + 16 + lm] = f2bf(pb);
    }
    __syncthreads();
    FragU pf;                         // reload P as one 16x32 A-fragment
    pf.q[0] = *(const u32x4*)&Ps[lm * 32 + bA];
    pf.q[1] = *(const u32x4*)&Ps[lm * 32 + bA + 8];
    __syncthreads();
    // ---- O += P V : V^T layout makes B-fragments contiguous (8 WMMAs) ----
#pragma unroll
    for (int d = 0; d < 8; ++d) {
      const unsigned short* vp = Vb + (size_t)(d * 16 + lm) * S + kb + kB;
      FragU vf; vf.q[0] = *(const u32x4*)vp; vf.q[1] = *(const u32x4*)(vp + 8);
      o[d] = wmma_bf16(pf.v, vf.v, o[d]);
    }
  }
  // ---- normalize and scatter into ctx[s][h*128+d] (bf16) ----
#pragma unroll
  for (int g = 0; g < 8; ++g) {
    const int m = q0 + g + 8 * lh;
    const float inv = 1.0f / lrow[g];
#pragma unroll
    for (int d = 0; d < 8; ++d) {
      ctx[(size_t)m * DM + h * HD + d * 16 + lm] = f2bf(o[d][g] * inv);
    }
  }
}

// ------------------------------ launcher -----------------------------------
extern "C" void kernel_launch(void* const* d_in, const int* in_sizes, int n_in,
                              void* d_out, int out_size, void* d_ws,
                              size_t ws_size, hipStream_t stream) {
  const float* x  = (const float*)d_in[0];
  const float* Wq = (const float*)d_in[1];
  const float* Wk = (const float*)d_in[2];
  const float* Wv = (const float*)d_in[3];
  const float* Wo = (const float*)d_in[4];

  char* ws = (char*)d_ws;
  size_t off = 0;
  auto carve = [&](size_t bytes) -> char* {
    char* p = ws + off;
    off += (bytes + 255) & ~(size_t)255;
    return p;
  };
  unsigned short* xb  = (unsigned short*)carve((size_t)S * DM * 2);
  unsigned short* Wqb = (unsigned short*)carve((size_t)DM * DM * 2);
  unsigned short* Wkb = (unsigned short*)carve((size_t)DM * DM * 2);
  unsigned short* Wvb = (unsigned short*)carve((size_t)DM * DM * 2);
  unsigned short* Wob = (unsigned short*)carve((size_t)DM * DM * 2);
  unsigned short* Qhp = (unsigned short*)carve((size_t)H * S * HD * 2);
  unsigned short* Khp = (unsigned short*)carve((size_t)H * S * HD * 2);
  unsigned short* Vtp = (unsigned short*)carve((size_t)H * HD * S * 2);
  unsigned short* ctx = (unsigned short*)carve((size_t)S * DM * 2);
  (void)ws_size; (void)in_sizes; (void)n_in; (void)out_size;

  cast_f32_bf16<<<(S * DM) / 256, 256, 0, stream>>>(x, xb, S * DM);
  cast_f32_bf16<<<(DM * DM) / 256, 256, 0, stream>>>(Wq, Wqb, DM * DM);
  cast_f32_bf16<<<(DM * DM) / 256, 256, 0, stream>>>(Wk, Wkb, DM * DM);
  cast_f32_bf16<<<(DM * DM) / 256, 256, 0, stream>>>(Wv, Wvb, DM * DM);
  cast_f32_bf16<<<(DM * DM) / 256, 256, 0, stream>>>(Wo, Wob, DM * DM);

  dim3 ggrid(S / 64, DM / 64);
  gemm_bt_wmma<<<ggrid, 128, 0, stream>>>(xb, Wqb, Qhp, 0);
  gemm_bt_wmma<<<ggrid, 128, 0, stream>>>(xb, Wkb, Khp, 0);
  gemm_bt_wmma<<<ggrid, 128, 0, stream>>>(xb, Wvb, Vtp, 1);

  rope_kernel<<<(H * S * 64) / 256, 256, 0, stream>>>(Qhp, Khp);

  attn_kernel<<<H * (S / 16), 32, 0, stream>>>(Qhp, Khp, Vtp, ctx);

  gemm_bt_wmma<<<ggrid, 128, 0, stream>>>(ctx, Wob, (float*)d_out, 2);
}
